// AttentionHead_67405216743502
// MI455X (gfx1250) — compile-verified, matrix operand order checked by reading, not run on hardware
//
#include <hip/hip_runtime.h>

#define B_ 4
#define S_ 4096
#define D_ 512

typedef __attribute__((ext_vector_type(16))) __bf16          bf16x16;
typedef __attribute__((ext_vector_type(8)))  float           f32x8;
typedef __attribute__((ext_vector_type(4)))  float           f32x4;
typedef __attribute__((ext_vector_type(8)))  unsigned short  u16x8;

// Native f32 -> bf16 convert (RNE); lets the backend use v_cvt_*bf16* ops
// instead of a scalarized shift/round sequence.
static __device__ __forceinline__ unsigned short f2bf(float f) {
    __bf16 h = (__bf16)f;
    return __builtin_bit_cast(unsigned short, h);
}

union FragU { u16x8 h[2]; bf16x16 b; };

// Load a bf16 A/B fragment: two contiguous 8-element runs at p and p+16,
// matching the 16-bit WMMA operand layout (K 0..7 & 16..23 per lane-half).
static __device__ __forceinline__ bf16x16 load_frag(const unsigned short* p) {
    FragU u;
    u.h[0] = *(const u16x8*)(p);
    u.h[1] = *(const u16x8*)(p + 16);
    return u.b;
}

static __device__ __forceinline__ f32x8 wmma_bf16(bf16x16 a, bf16x16 b, f32x8 c) {
    return __builtin_amdgcn_wmma_f32_16x16x32_bf16(false, a, false, b, (short)0, c, false, false);
}

// ---------------------------------------------------------------------------
// Kernel 0: convert the three weight matrices to bf16 (row-major [D][D]).
// ---------------------------------------------------------------------------
__global__ void conv_w_kernel(const float* __restrict__ wk,
                              const float* __restrict__ wq,
                              const float* __restrict__ wv,
                              unsigned short* __restrict__ wkb,
                              unsigned short* __restrict__ wqb,
                              unsigned short* __restrict__ wvb) {
    int i = blockIdx.x * blockDim.x + threadIdx.x;
    if (i < D_ * D_) {
        wkb[i] = f2bf(wk[i]);
        wqb[i] = f2bf(wq[i]);
        wvb[i] = f2bf(wv[i]);
    }
}

// ---------------------------------------------------------------------------
// Kernel 1: fused K/Q/V projection.  out[s,d] = sum_k x[s,k] * W[d,k]
// One wave (32 threads) per 16x16 tile; shares the x A-fragment across the
// three weight B-fragments (3 WMMAs per K-step of 32).
// K,Q stored row-major bf16 [B][S][D]; V stored transposed bf16 [B][D][S].
// ---------------------------------------------------------------------------
__global__ __launch_bounds__(32)
void proj_kqv_kernel(const float* __restrict__ x,
                     const unsigned short* __restrict__ wkb,
                     const unsigned short* __restrict__ wqb,
                     const unsigned short* __restrict__ wvb,
                     unsigned short* __restrict__ kws,
                     unsigned short* __restrict__ qws,
                     unsigned short* __restrict__ vtws) {
    const int lane = threadIdx.x;
    const int l15  = lane & 15;
    const int half = lane >> 4;
    const int d0 = blockIdx.x * 16;
    const int s0 = blockIdx.y * 16;
    const int b  = blockIdx.z;

    const f32x8 vzero = {};
    f32x8 ak = vzero, aq = vzero, av = vzero;

    const float* xrow = x + ((size_t)(b * S_) + s0 + l15) * D_;

    for (int kk = 0; kk < D_; kk += 32) {
        // A fragment: row s0+l15 of x, f32 -> bf16 via native converts
        const float* pa = xrow + kk + half * 8;
        f32x4 a0 = *(const f32x4*)(pa + 0);
        f32x4 a1 = *(const f32x4*)(pa + 4);
        f32x4 a2 = *(const f32x4*)(pa + 16);
        f32x4 a3 = *(const f32x4*)(pa + 20);
        bf16x16 afrag;
        #pragma unroll
        for (int i = 0; i < 4; ++i) {
            afrag[i]      = (__bf16)a0[i];
            afrag[4 + i]  = (__bf16)a1[i];
            afrag[8 + i]  = (__bf16)a2[i];
            afrag[12 + i] = (__bf16)a3[i];
        }

        // B fragments: W rows are contiguous over k
        const size_t woff = (size_t)(d0 + l15) * D_ + kk + half * 8;
        bf16x16 bk = load_frag(wkb + woff);
        bf16x16 bq = load_frag(wqb + woff);
        bf16x16 bv = load_frag(wvb + woff);

        ak = wmma_bf16(afrag, bk, ak);
        aq = wmma_bf16(afrag, bq, aq);
        av = wmma_bf16(afrag, bv, av);
    }

    #pragma unroll
    for (int e = 0; e < 8; ++e) {
        const int row = s0 + e + half * 8;
        const int col = d0 + l15;
        kws [((size_t)(b * S_) + row) * D_ + col] = f2bf(ak[e]);
        qws [((size_t)(b * S_) + row) * D_ + col] = f2bf(aq[e]);
        vtws[((size_t)(b * D_) + col) * S_ + row] = f2bf(av[e]);   // transposed
    }
}

// ---------------------------------------------------------------------------
// Kernel 2: flash attention with swapped roles:
//   scores[s,t] = <keys[s], queries[t]> * scale ; mask & softmax over t
//   out[s,:]    = sum_t P[s,t] * values[t,:]
// One wave owns 16 rows (s0..s0+15); keys rows staged once into LDS via the
// CDNA5 async global->LDS pipe; t streamed in chunks of 32 (two score tiles
// -> one 16x32 P A-fragment -> 32 PV WMMAs).  Full 16x512 f32 output tile
// accumulated in registers (32 x v8f).
// ---------------------------------------------------------------------------
__global__ __launch_bounds__(32)
void flash_attn_kernel(const unsigned short* __restrict__ kws,
                       const unsigned short* __restrict__ qws,
                       const unsigned short* __restrict__ vtws,
                       const int* __restrict__ mask,
                       float* __restrict__ out) {
    __shared__ __align__(16) unsigned short smK[16 * D_];   // 16 KB key rows
    __shared__ __align__(16) unsigned short smP[16 * 32];   // 1 KB P tile

    const int lane = threadIdx.x;
    const int l15  = lane & 15;
    const int half = lane >> 4;
    const int s0 = blockIdx.x * 16;
    const int b  = blockIdx.y;
    const float scale = 0.044194173824159216f;  // 1/sqrt(512)

    // ---- stage this block's 16 key rows (16 KB) into LDS asynchronously ----
    // generic->LDS pointer truncation recovers the wave-relative LDS offset.
    {
        const unsigned short* ksrc = kws + ((size_t)(b * S_) + s0) * D_;
        unsigned      ldsoff = (unsigned)(uintptr_t)(&smK[0]) + (unsigned)lane * 16u;
        unsigned long gaddr  = (unsigned long)(uintptr_t)ksrc + (unsigned long)lane * 16u;
        #pragma unroll 4
        for (int i = 0; i < 32; ++i) {   // 32 lanes x 16B x 32 iters = 16 KB
            asm volatile("global_load_async_to_lds_b128 %0, %1, off"
                         :: "v"(ldsoff + (unsigned)i * 512u),
                            "v"(gaddr + (unsigned long)i * 512u)
                         : "memory");
        }
        asm volatile("s_wait_asynccnt 0" ::: "memory");
    }
    __syncthreads();

    const f32x8 vzero = {};
    f32x8 o[32];
    #pragma unroll
    for (int t = 0; t < 32; ++t) o[t] = vzero;

    float m_i[8], l_i[8], rsc[8];
    #pragma unroll
    for (int e = 0; e < 8; ++e) { m_i[e] = -3.0e38f; l_i[e] = 0.0f; }

    const int* mrow = mask + b * S_;
    const unsigned short* kbase = smK + l15 * D_ + half * 8;

    for (int t0 = 0; t0 < S_; t0 += 32) {
        if (t0 + 32 < S_) {  // hint next chunk of queries toward the caches
            __builtin_prefetch(qws + ((size_t)(b * S_) + t0 + 32 + l15) * D_, 0, 1);
        }

        // ---- score tiles: rows = keys, cols = queries t0..t0+31 ----
        f32x8 sc0 = vzero, sc1 = vzero;
        const unsigned short* q0 = qws + ((size_t)(b * S_) + t0 + l15) * D_ + half * 8;
        const unsigned short* q1 = q0 + 16 * D_;
        for (int kk = 0; kk < D_; kk += 32) {
            bf16x16 afrag = load_frag(kbase + kk);
            bf16x16 b0 = load_frag(q0 + kk);
            bf16x16 b1 = load_frag(q1 + kk);
            sc0 = wmma_bf16(afrag, b0, sc0);
            sc1 = wmma_bf16(afrag, b1, sc1);
        }

        // ---- scale + column mask (mask indexes t = column) ----
        const float neg = -1.0e12f;
        const bool msk0 = (mrow[t0 + l15] != 0);
        const bool msk1 = (mrow[t0 + 16 + l15] != 0);
        float p0[8], p1[8];
        #pragma unroll
        for (int e = 0; e < 8; ++e) {
            p0[e] = msk0 ? sc0[e] * scale : neg;
            p1[e] = msk1 ? sc1[e] * scale : neg;
        }

        // ---- online softmax: row reductions across 16 lanes ----
        #pragma unroll
        for (int e = 0; e < 8; ++e) {
            float v = fmaxf(p0[e], p1[e]);
            v = fmaxf(v, __shfl_xor(v, 1, 32));
            v = fmaxf(v, __shfl_xor(v, 2, 32));
            v = fmaxf(v, __shfl_xor(v, 4, 32));
            v = fmaxf(v, __shfl_xor(v, 8, 32));
            const float mnew = fmaxf(m_i[e], v);
            const float r  = __expf(m_i[e] - mnew);
            const float e0 = __expf(p0[e] - mnew);
            const float e1 = __expf(p1[e] - mnew);
            p0[e] = e0; p1[e] = e1;
            float s = e0 + e1;
            s += __shfl_xor(s, 1, 32);
            s += __shfl_xor(s, 2, 32);
            s += __shfl_xor(s, 4, 32);
            s += __shfl_xor(s, 8, 32);
            l_i[e] = l_i[e] * r + s;
            m_i[e] = mnew;
            rsc[e] = r;
        }

        // ---- rescale running output ----
        #pragma unroll
        for (int t = 0; t < 32; ++t) {
            #pragma unroll
            for (int e = 0; e < 8; ++e) o[t][e] *= rsc[e];
        }

        // ---- P tile -> LDS (bf16, [16 rows][32 t-cols]) -> A fragment ----
        #pragma unroll
        for (int e = 0; e < 8; ++e) {
            const int row = e + half * 8;
            smP[row * 32 + l15]      = f2bf(p0[e]);
            smP[row * 32 + 16 + l15] = f2bf(p1[e]);
        }
        __syncthreads();
        bf16x16 pfrag = load_frag(smP + l15 * 32 + half * 8);

        // ---- P (16x32) x V (32x512): 32 WMMAs, B-frags from transposed V ----
        const unsigned short* vb = vtws + ((size_t)(b * D_) + l15) * S_ + t0 + half * 8;
        #pragma unroll
        for (int t = 0; t < 32; ++t) {
            bf16x16 bfrag = load_frag(vb + (size_t)(t * 16) * S_);
            o[t] = wmma_bf16(pfrag, bfrag, o[t]);
        }
        __syncthreads();
    }

    // ---- normalize and store ----
    #pragma unroll
    for (int e = 0; e < 8; ++e) {
        const float li = l_i[e];
        l_i[e] = (li > 0.0f) ? (1.0f / li) : 0.0f;
    }
    #pragma unroll
    for (int t = 0; t < 32; ++t) {
        #pragma unroll
        for (int e = 0; e < 8; ++e) {
            out[((size_t)(b * S_) + s0 + e + half * 8) * D_ + t * 16 + l15] =
                o[t][e] * l_i[e];
        }
    }
}

// ---------------------------------------------------------------------------
extern "C" void kernel_launch(void* const* d_in, const int* in_sizes, int n_in,
                              void* d_out, int out_size, void* d_ws, size_t ws_size,
                              hipStream_t stream) {
    const float* x    = (const float*)d_in[0];
    const int*   mask = (const int*)d_in[1];
    const float* Wk   = (const float*)d_in[2];
    const float* Wq   = (const float*)d_in[3];
    const float* Wv   = (const float*)d_in[4];
    float* out = (float*)d_out;

    // Workspace layout (bf16 elements): K, Q, V^T, then 3 weight matrices.
    const size_t nKQ = (size_t)B_ * S_ * D_;          // 8 Mi elements each
    unsigned short* kws  = (unsigned short*)d_ws;
    unsigned short* qws  = kws + nKQ;
    unsigned short* vtws = qws + nKQ;
    unsigned short* wkb  = vtws + nKQ;
    unsigned short* wqb  = wkb + (size_t)D_ * D_;
    unsigned short* wvb  = wqb + (size_t)D_ * D_;

    conv_w_kernel<<<dim3((D_ * D_ + 255) / 256), dim3(256), 0, stream>>>(
        Wk, Wq, Wv, wkb, wqb, wvb);

    proj_kqv_kernel<<<dim3(D_ / 16, S_ / 16, B_), dim3(32), 0, stream>>>(
        x, wkb, wqb, wvb, kws, qws, vtws);

    flash_attn_kernel<<<dim3(S_ / 16, B_), dim3(32), 0, stream>>>(
        kws, qws, vtws, mask, out);
}
